// MyLSTM2_4grps_28406913696413
// MI455X (gfx1250) — compile-verified, matrix operand order checked by reading, not run on hardware
//
#include <hip/hip_runtime.h>
#include <hip/hip_bf16.h>
#include <math.h>

// MI455X / gfx1250: wave32, WMMA 16x16x32 f16 -> f32.
typedef __attribute__((ext_vector_type(16))) _Float16 v16h;
typedef __attribute__((ext_vector_type(8)))  float    v8f;

#define BATCH   128
#define TSTEPS  256
#define HDIM    256
#define NCOLS   1024        // 4*H gate columns
#define KT_MAX  11          // max (Fpad+256)/32  (LSTM1: 96+256 -> 11)
#define NTHR    512         // 16 waves per workgroup

// CDNA5 16-bit A-matrix 16x32 lane layout (ISA 7.12.2):
// lane group 0-15 holds K {0..7,16..23}, group 16-31 holds K {8..15,24..31},
// element e in the v16h maps to k = (e&7) + ((e>>3)<<4) + half*8.
__device__ __forceinline__ int kmap(int e, int half) {
  return (e & 7) + ((e >> 3) << 4) + (half << 3);
}
__device__ __forceinline__ float sigm(float x) { return 1.0f / (1.0f + __expf(-x)); }

// ---------------------------------------------------------------------------
// Pre-swizzle [W(F x 1024, zero-padded to Fpad) ; U(256 x 1024)] (f32, row
// major) into f16 WMMA B-tile blocks: layout [ntile(64)][kt(KT)][lane(32)][e(16)].
// Each lane's 16 f16 (32B) are contiguous -> b128-pair loads in the hot loop.
// ---------------------------------------------------------------------------
__global__ void swizzle_wu_kernel(const float* __restrict__ W,
                                  const float* __restrict__ U,
                                  _Float16* __restrict__ Bm,
                                  int F, int Fpad, int KT) {
  int total = 64 * KT * 512;
  for (int i = blockIdx.x * blockDim.x + threadIdx.x; i < total;
       i += gridDim.x * blockDim.x) {
    int ntile = i / (KT * 512);
    int rem   = i - ntile * (KT * 512);
    int kt = rem >> 9;
    int r2 = rem & 511;
    int ln = r2 >> 4;
    int e  = r2 & 15;
    int k  = kt * 32 + kmap(e, ln >> 4);
    int n  = (ntile << 4) + (ln & 15);
    float v = 0.0f;
    if (k < Fpad) {
      if (k < F) v = W[k * NCOLS + n];
    } else {
      v = U[(k - Fpad) * NCOLS + n];
    }
    Bm[i] = (_Float16)v;
  }
}

// ---------------------------------------------------------------------------
// Templated LSTM recurrence body. 16 waves; wave w owns gate columns
// [w*16, w*16+16) of each of the four gates (4 WMMA accumulator tiles).
// Compile-time K-tile count -> fully unrolled SSA accumulator chain.
// `zoff` is 0 at runtime but opaque: defeats LICM hoist/spill of B loads.
// ---------------------------------------------------------------------------
template <int F, int FPAD, int KT>
__device__ __forceinline__ void lstm_body(
    const float* __restrict__ x,      // [B,T,F]
    const float* __restrict__ bias,   // [4H]
    const _Float16* __restrict__ Bm,  // swizzled [W;U]
    float* __restrict__ merged,       // [B,T,1024]
    int lstm, int row_base, int zoff,
    _Float16* abuf, float* maskv) {
  constexpr int XT = FPAD / 32;       // x K-tiles; h K-tiles are fixed 8
  constexpr int NT = KT * 4;          // B tiles per wave per step
  constexpr int PF = 4;               // B prefetch pipeline depth
  const int tid  = threadIdx.x;
  const int w    = tid >> 5;          // 0..15
  const int lane = tid & 31;
  const int half = lane >> 4;
  const int c15  = lane & 15;

  // Per-lane gate biases (column = w*16 + c15 within each gate).
  float bias_r[4];
#pragma unroll
  for (int g = 0; g < 4; ++g)
    bias_r[g] = bias[(g << 8) + (w << 4) + c15];

  // Cell state in VGPRs: C-matrix layout (VGPR r, lane) = (row r+8*half, col c15).
  float creg[8];
#pragma unroll
  for (int r = 0; r < 8; ++r) creg[r] = 0.0f;

  // h0 = 0 in the swizzled h region.
  for (int i = tid; i < (8 << 9); i += NTHR) abuf[(XT << 9) + i] = (_Float16)0.0f;

  const int t2      = w & 1;                // half of the h K-tile this wave owns
  const int hbase   = (XT + (w >> 1)) << 9; // wave pair (w>>1) owns h K-tile
  const int lane_hi = (c15 >> 3) << 4;      // A-layout lane group for its column

  for (int t = 0; t < TSTEPS; ++t) {
    // Opaque zero offset: defeats LICM hoisting of the (t-invariant) B loads.
    const _Float16* Bmt = Bm + (size_t)zoff * (size_t)t;

    // ---- Phase 1: stage x_t (swizzled, zero-padded) + compute mask --------
#pragma unroll
    for (int i = tid; i < (XT << 9); i += NTHR) {
      int kt = i >> 9;
      int r2 = i & 511;
      int ln = r2 >> 4;
      int e  = r2 & 15;
      int k  = kt * 32 + kmap(e, ln >> 4);
      int m  = ln & 15;
      float v = 0.0f;
      if (k < F)
        v = x[((size_t)(row_base + m) * TSTEPS + t) * F + k];
      abuf[i] = (_Float16)v;
    }
    if (w == 0 && lane < 16) {
      const float* xr = x + ((size_t)(row_base + lane) * TSTEPS + t) * F;
      int any = 0;
#pragma unroll 4
      for (int k = 0; k < F; ++k) any |= (xr[k] != -1.0f);
      maskv[lane] = any ? 1.0f : 0.0f;
    }
    __syncthreads();   // x + mask ready; previous step's h writes visible

    // ---- Phase 2: Z = [x_t , h] * [W;U] + b  (4 WMMA tiles per wave) -------
    v8f z[4];
#pragma unroll
    for (int g = 0; g < 4; ++g) {
      v8f zz;
#pragma unroll
      for (int r = 0; r < 8; ++r) zz[r] = bias_r[g];
      z[g] = zz;
    }
    // 4-deep rotating B prefetch buffer: load tile i+PF before consuming i.
    v16h bb[PF];
#pragma unroll
    for (int i = 0; i < PF; ++i) {
      int kt = i >> 2, g = i & 3;
      bb[i] = *(const v16h*)(Bmt +
               (((size_t)((((g << 4) + w) * KT) + kt)) << 9) + (lane << 4));
    }
#pragma unroll
    for (int kt = 0; kt < KT; ++kt) {
      v16h A = *(const v16h*)(abuf + (kt << 9) + (lane << 4));
#pragma unroll
      for (int g = 0; g < 4; ++g) {
        const int idx = kt * 4 + g;
        v16h bcur = bb[idx & (PF - 1)];
        if (idx + PF < NT) {
          const int nkt = (idx + PF) >> 2, ng = (idx + PF) & 3;
          bb[idx & (PF - 1)] = *(const v16h*)(Bmt +
              (((size_t)((((ng << 4) + w) * KT) + nkt)) << 9) + (lane << 4));
        }
        z[g] = __builtin_amdgcn_wmma_f32_16x16x32_f16(
            false, A, false, bcur, (short)0, z[g], false, false);
      }
    }

    // ---- Phase 3: gates, cell update, masking ------------------------------
    float hnew[8];
#pragma unroll
    for (int r = 0; r < 8; ++r) {
      int m = r + (half << 3);
      float iv = sigm(z[0][r]);
      float fv = sigm(z[1][r]);
      float gv = fmaxf(z[2][r], 0.0f);     // cell activation = relu
      float ov = sigm(z[3][r]);
      float cn = fv * creg[r] + iv * gv;
      float hv = ov * fmaxf(cn, 0.0f);     // output activation = relu
      float mk = maskv[m];
      int aidx = hbase + ((m + lane_hi) << 4) + (c15 & 7) + (t2 << 3);
      float hp = (float)abuf[aidx];        // previous h (for masked rows)
      creg[r] = mk * cn + (1.0f - mk) * creg[r];
      hnew[r] = mk * hv + (1.0f - mk) * hp;
    }
    __syncthreads();   // everyone done reading h/x for this step

    // ---- Phase 4: write h back (swizzled LDS) + merged output --------------
#pragma unroll
    for (int r = 0; r < 8; ++r) {
      int m = r + (half << 3);
      int aidx = hbase + ((m + lane_hi) << 4) + (c15 & 7) + (t2 << 3);
      abuf[aidx] = (_Float16)hnew[r];
      int ncol = (w << 4) + c15;
      merged[((size_t)(row_base + m) * TSTEPS + t) * NCOLS +
             (lstm << 8) + ncol] = hnew[r];
    }
  }
}

struct LstmP {
  const float*    x;
  const float*    bias;
  const _Float16* Bm;
};
struct AllP {
  LstmP  p[4];
  float* merged;
  int    zoff;   // always 0 at runtime; opaque to the compiler
};

__global__ void __launch_bounds__(NTHR)
lstm_fused_kernel(AllP P) {
  __shared__ alignas(32) _Float16 abuf[KT_MAX * 512];
  __shared__ float maskv[16];
  const int lstm     = blockIdx.y;
  const int row_base = blockIdx.x << 4;
  switch (lstm) {   // uniform per y-slice; four compile-time specializations
    case 0:
      lstm_body<80, 96, 11>(P.p[0].x, P.p[0].bias, P.p[0].Bm, P.merged,
                            0, row_base, P.zoff, abuf, maskv);
      break;
    case 1:
      lstm_body<64, 64, 10>(P.p[1].x, P.p[1].bias, P.p[1].Bm, P.merged,
                            1, row_base, P.zoff, abuf, maskv);
      break;
    case 2:
      lstm_body<48, 64, 10>(P.p[2].x, P.p[2].bias, P.p[2].Bm, P.merged,
                            2, row_base, P.zoff, abuf, maskv);
      break;
    default:
      lstm_body<32, 32, 9>(P.p[3].x, P.p[3].bias, P.p[3].Bm, P.merged,
                           3, row_base, P.zoff, abuf, maskv);
      break;
  }
}

// ---------------------------------------------------------------------------
// Attention + output head: one block per batch row.
// ---------------------------------------------------------------------------
__global__ void __launch_bounds__(256)
attn_kernel(const float* __restrict__ merged,
            const float* __restrict__ w_att, const float* __restrict__ b_att,
            const float* __restrict__ w_out, const float* __restrict__ b_out,
            float* __restrict__ out) {
  const int b    = blockIdx.x;
  const int tid  = threadIdx.x;
  const int w    = tid >> 5;
  const int lane = tid & 31;
  __shared__ float sc[TSTEPS];
  __shared__ float red[256];
  const float* mb = merged + (size_t)b * TSTEPS * NCOLS;

  // scores (one wave per timestep, wave-shuffle reduction)
  for (int t = w; t < TSTEPS; t += 8) {
    const float* row = mb + (size_t)t * NCOLS;
    float p = 0.0f;
    for (int k = lane; k < NCOLS; k += 32) p += row[k] * w_att[k];
    for (int off = 16; off; off >>= 1) p += __shfl_xor(p, off, 32);
    if (lane == 0) sc[t] = tanhf(p + b_att[0]);
  }
  __syncthreads();

  // softmax over T=256
  float v = sc[tid];
  red[tid] = v;
  __syncthreads();
  for (int s = 128; s; s >>= 1) {
    if (tid < s) red[tid] = fmaxf(red[tid], red[tid + s]);
    __syncthreads();
  }
  float mx = red[0];
  __syncthreads();
  float e = __expf(v - mx);
  red[tid] = e;
  __syncthreads();
  for (int s = 128; s; s >>= 1) {
    if (tid < s) red[tid] += red[tid + s];
    __syncthreads();
  }
  float inv = 1.0f / red[0];
  __syncthreads();
  sc[tid] = e * inv;
  __syncthreads();

  // rep = sum_t attn[t]*merged[b,t,:]; thread owns cols {tid, tid+256, ...}
  float acc[4] = {0.0f, 0.0f, 0.0f, 0.0f};
  for (int t = 0; t < TSTEPS; ++t) {
    float a = sc[t];
    const float* row = mb + (size_t)t * NCOLS;
#pragma unroll
    for (int j = 0; j < 4; ++j) acc[j] += a * row[tid + (j << 8)];
  }
  float po = 0.0f;
#pragma unroll
  for (int j = 0; j < 4; ++j) po += acc[j] * w_out[tid + (j << 8)];
  red[tid] = po;
  __syncthreads();
  for (int s = 128; s; s >>= 1) {
    if (tid < s) red[tid] += red[tid + s];
    __syncthreads();
  }
  if (tid == 0) out[b] = 1.0f / (1.0f + __expf(-(red[0] + b_out[0])));
}

// ---------------------------------------------------------------------------
extern "C" void kernel_launch(void* const* d_in, const int* in_sizes, int n_in,
                              void* d_out, int out_size, void* d_ws, size_t ws_size,
                              hipStream_t stream) {
  (void)in_sizes; (void)n_in; (void)out_size; (void)ws_size;

  // setup_inputs() dict order: x1,W1,U1,b1, x2,W2,U2,b2, x3,W3,U3,b3,
  // x4,W4,U4,b4, w_att,b_att,w_out,b_out  (all float32)
  const float* x[4];
  const float* W[4];
  const float* U[4];
  const float* bias[4];
  for (int i = 0; i < 4; ++i) {
    x[i]    = (const float*)d_in[4 * i + 0];
    W[i]    = (const float*)d_in[4 * i + 1];
    U[i]    = (const float*)d_in[4 * i + 2];
    bias[i] = (const float*)d_in[4 * i + 3];
  }
  const float* w_att = (const float*)d_in[16];
  const float* b_att = (const float*)d_in[17];
  const float* w_out = (const float*)d_in[18];
  const float* b_out = (const float*)d_in[19];

  static const int Farr[4]    = {80, 64, 48, 32};
  static const int FpadArr[4] = {96, 64, 64, 32};

  // Workspace layout: merged [B,T,1024] f32, then per-LSTM swizzled B-tiles.
  char*  ws     = (char*)d_ws;
  float* merged = (float*)ws;
  size_t off    = (size_t)BATCH * TSTEPS * NCOLS * sizeof(float);

  AllP P;
  P.merged = merged;
  P.zoff   = 0;   // runtime zero, opaque to device compiler
  _Float16* Bm[4];
  int KT[4];
  for (int i = 0; i < 4; ++i) {
    KT[i]  = (FpadArr[i] + HDIM) / 32;
    Bm[i]  = (_Float16*)(ws + off);
    off   += (size_t)64 * KT[i] * 512 * sizeof(_Float16);
    P.p[i].x    = x[i];
    P.p[i].bias = bias[i];
    P.p[i].Bm   = Bm[i];
  }

  // 1) swizzle/convert weights into WMMA B-tile layout (f16, lives in L2)
  for (int i = 0; i < 4; ++i) {
    int total  = 64 * KT[i] * 512;
    int blocks = (total + 255) / 256;
    swizzle_wu_kernel<<<blocks, 256, 0, stream>>>(W[i], U[i], Bm[i],
                                                  Farr[i], FpadArr[i], KT[i]);
  }

  // 2) fused LSTM recurrence: 8 batch tiles x 4 LSTMs, 16 waves each
  lstm_fused_kernel<<<dim3(8, 4), NTHR, 0, stream>>>(P);

  // 3) attention + sigmoid head -> d_out [B,1] f32
  attn_kernel<<<BATCH, 256, 0, stream>>>(merged, w_att, b_att, w_out, b_out,
                                         (float*)d_out);
}
